// CausalMultiHeadAttention_67954972557390
// MI455X (gfx1250) — matrix operand order repaired and checked by reading, not compiled, not run on hardware
//
#include <hip/hip_runtime.h>
#include <hip/hip_bf16.h>

// ---------------------------------------------------------------------------
// CDNA5 (gfx1250) bf16 WMMA causal MHA.
//   v_wmma_f32_16x16x32_bf16 for all matrix ops; f32 accumulate; bf16
//   intermediates; DPP16 row reductions; async global->LDS K staging.
// ---------------------------------------------------------------------------

typedef __attribute__((ext_vector_type(16))) __bf16 v16bf;
typedef __attribute__((ext_vector_type(8)))  float  v8f;

struct U4 { unsigned int x, y, z, w; };

__device__ __forceinline__ unsigned short f32_to_bf16(float f) {
  unsigned int u = __float_as_uint(f);
  u += 0x7FFFu + ((u >> 16) & 1u);          // round-to-nearest-even
  return (unsigned short)(u >> 16);
}

union FragU {
  v16bf v;
  U4 q[2];
  unsigned short us[16];
};

// 16 contiguous bf16 = two 16B chunks -> one v16bf fragment register group.
__device__ __forceinline__ v16bf load_frag16(const unsigned short* p0,
                                             const unsigned short* p1) {
  FragU u;
  u.q[0] = *(const U4*)p0;
  u.q[1] = *(const U4*)p1;
  return u.v;
}

// ---- DPP16 cross-lane reductions within each 16-lane group (VALU only) ----
template <int CTRL>
__device__ __forceinline__ float dpp_f32(float x) {
  return __uint_as_float((unsigned)__builtin_amdgcn_update_dpp(
      0, (int)__float_as_uint(x), CTRL, 0xF, 0xF, true));
}
__device__ __forceinline__ float reduce_max16(float x) {
  x = fmaxf(x, dpp_f32<0xB1>(x));   // quad_perm [1,0,3,2] : xor 1
  x = fmaxf(x, dpp_f32<0x4E>(x));   // quad_perm [2,3,0,1] : xor 2
  x = fmaxf(x, dpp_f32<0x141>(x));  // row_half_mirror     : xor 7
  x = fmaxf(x, dpp_f32<0x140>(x));  // row_mirror          : xor 15
  return x;
}
__device__ __forceinline__ float reduce_sum16(float x) {
  x += dpp_f32<0xB1>(x);
  x += dpp_f32<0x4E>(x);
  x += dpp_f32<0x141>(x);
  x += dpp_f32<0x140>(x);
  return x;
}

// ---- CDNA5 async global->LDS copy (ASYNCcnt-tracked) ----------------------
// Low 32 bits of a generic pointer into LDS are the LDS byte offset.
__device__ __forceinline__ unsigned lds_off(const void* p) {
  return (unsigned)(unsigned long long)p;
}
__device__ __forceinline__ void async_load_b128(unsigned lds_byte_off,
                                                const void* gaddr) {
  asm volatile("global_load_async_to_lds_b128 %0, %1, off"
               :: "v"(lds_byte_off), "v"(gaddr) : "memory");
}
__device__ __forceinline__ void wait_asynccnt0() {
  asm volatile("s_wait_asynccnt 0" ::: "memory");
}

// ---------------------------------------------------------------------------
// GEMM:  C[M,N] = A[M,K] * B[K,N] + bias[N]
//   A: f32 (converted) or bf16 (ushort).  B: f32, converted on stage.
//   Block: 256 threads = 8 waves; C tile 64x64; wave tile 16x32.
//   K-step 32 -> one 16x32 A frag + two 32x16 B frags -> 2 WMMAs.
// ---------------------------------------------------------------------------
template <bool A_IS_F32, bool STORE_BF16>
__global__ __launch_bounds__(256)
void wmma_gemm_bias(const void* __restrict__ Av,
                    const float* __restrict__ B,
                    const float* __restrict__ bias,
                    void* __restrict__ Cv,
                    int M, int N, int K, int lda, int ldb, int ldc) {
  constexpr int LP = 40;  // LDS pitch (ushorts): 32 + 8 pad, keeps 16B align
  __shared__ __align__(16) unsigned short ldsA[64 * LP];   // 64 rows x 32 k
  __shared__ __align__(16) unsigned short ldsBt[64 * LP];  // 64 n   x 32 k (transposed)

  const int tid   = threadIdx.x;
  const int lane  = tid & 31;
  const int wave  = tid >> 5;     // 0..7
  const int waveM = wave & 3;     // 0..3  (M sub-tile)
  const int waveN = wave >> 2;    // 0..1  (N sub-tile, 32 wide)
  const int mBase = blockIdx.y * 64;
  const int nBase = blockIdx.x * 64;
  const int half  = lane >> 4;    // lane half selects K interleave
  const int l16   = lane & 15;

  v8f c0 = {};
  v8f c1 = {};

  const float*          Af = (const float*)Av;
  const unsigned short* Ab = (const unsigned short*)Av;

  for (int kb = 0; kb < K; kb += 32) {
    // --- stage A tile (64x32) as bf16, row-major ---
    {
      const int row = tid >> 2;
      const int col = (tid & 3) * 8;
      const size_t g = (size_t)(mBase + row) * lda + kb + col;
      unsigned short* d = &ldsA[row * LP + col];
      if constexpr (A_IS_F32) {
        unsigned int* d32 = (unsigned int*)d;
#pragma unroll
        for (int i = 0; i < 4; ++i) {
          const unsigned lo = f32_to_bf16(Af[g + 2 * i]);
          const unsigned hi = f32_to_bf16(Af[g + 2 * i + 1]);
          d32[i] = lo | (hi << 16);
        }
        if (kb + 32 < K) __builtin_prefetch(&Af[g + 32], 0, 1);
      } else {
        ((U4*)d)[0] = *(const U4*)&Ab[g];
        ((U4*)d)[1] = *(const U4*)&Ab[g + 8];
        if (kb + 32 < K) __builtin_prefetch(&Ab[g + 32], 0, 1);
      }
    }
    // --- stage B tile (32x64) transposed -> ldsBt[n][k] ---
    {
      const int krow = tid >> 3;
      const int col  = (tid & 7) * 8;
      const size_t g = (size_t)(kb + krow) * ldb + nBase + col;
#pragma unroll
      for (int i = 0; i < 8; ++i)
        ldsBt[(col + i) * LP + krow] = f32_to_bf16(B[g + i]);
      if (kb + 32 < K) __builtin_prefetch(&B[g + (size_t)32 * ldb], 0, 1);
    }
    __syncthreads();

    // A frag: lane half 0 -> K 0..15; half 1 -> K 16..31
    const int arow = waveM * 16 + l16;
    const int aoff = half * 16;
    v16bf a = load_frag16(&ldsA[arow * LP + aoff],
                          &ldsA[arow * LP + 8 + aoff]);

    // B frags: lane = N column, 16 contiguous K (half 0: K0..15, half 1: K16..31)
    const int kst   = half * 16;
    const int bcol0 = waveN * 32 + l16;
    v16bf b0 = load_frag16(&ldsBt[bcol0 * LP + kst],
                           &ldsBt[bcol0 * LP + kst + 8]);
    v16bf b1 = load_frag16(&ldsBt[(bcol0 + 16) * LP + kst],
                           &ldsBt[(bcol0 + 16) * LP + kst + 8]);

    c0 = __builtin_amdgcn_wmma_f32_16x16x32_bf16(false, a, false, b0,
                                                 (short)0, c0, false, false);
    c1 = __builtin_amdgcn_wmma_f32_16x16x32_bf16(false, a, false, b1,
                                                 (short)0, c1, false, false);
    __syncthreads();
  }

  // Epilogue: C layout -> VGPR r: rows r / r+8 per lane half, col = lane%16.
  const int n0 = nBase + waveN * 32 + l16;
  const int n1 = n0 + 16;
  const float bias0 = bias[n0];
  const float bias1 = bias[n1];
#pragma unroll
  for (int r = 0; r < 8; ++r) {
    const int m = mBase + waveM * 16 + r + half * 8;
    const float v0 = c0[r] + bias0;
    const float v1 = c1[r] + bias1;
    if constexpr (STORE_BF16) {
      unsigned short* C = (unsigned short*)Cv;
      C[(size_t)m * ldc + n0] = f32_to_bf16(v0);
      C[(size_t)m * ldc + n1] = f32_to_bf16(v1);
    } else {
      float* C = (float*)Cv;
      C[(size_t)m * ldc + n0] = v0;
      C[(size_t)m * ldc + n1] = v1;
    }
  }
}

// ---------------------------------------------------------------------------
// Flash attention (causal), bf16 QKV in, bf16 out.
//   Block: 128 threads = 4 waves; block owns 64 query rows of one (b,h).
//   Wave owns 16 q rows. Key blocks of 32: K staged to LDS via
//   global_load_async_to_lds_b128, V staged transposed via regular DS.
//   Per key block per wave: 4 WMMAs for scores (D=64) + 4 WMMAs for P*V.
// ---------------------------------------------------------------------------
__global__ __launch_bounds__(128)
void flash_attn_kernel(const unsigned short* __restrict__ qkv,
                       unsigned short* __restrict__ attnOut) {
  constexpr int S = 1024, E = 1024, D = 64, LDQ = 3072;
  constexpr int KP = 72;  // K tile pitch (64 + 8), 16B-aligned rows
  constexpr int VP = 40;  // Vt / P pitch (32 + 8)
  __shared__ __align__(16) unsigned short ldsK[32 * KP];   // [key][d]
  __shared__ __align__(16) unsigned short ldsVt[64 * VP];  // [d][key]
  __shared__ __align__(16) unsigned short ldsP[4][16 * VP];

  const int tid  = threadIdx.x;
  const int lane = tid & 31;
  const int wave = tid >> 5;
  const int half = lane >> 4;
  const int l16  = lane & 15;

  const int qBlock    = blockIdx.x * 64;
  const int h         = blockIdx.y;
  const int b         = blockIdx.z;
  const int qWaveBase = qBlock + wave * 16;
  const int qWaveMax  = qWaveBase + 15;
  const size_t rowBase = (size_t)b * S;

  // Q fragments for both D-halves (A-matrix layout, read straight from global bf16)
  const unsigned short* qptr =
      qkv + (rowBase + qWaveBase + l16) * LDQ + h * D;
  const int aoff = half * 16;
  const v16bf qa0 = load_frag16(qptr + aoff,      qptr + 8 + aoff);
  const v16bf qa1 = load_frag16(qptr + 32 + aoff, qptr + 40 + aoff);

  float mrow[8], lrow[8];
  v8f o0 = {}, o1 = {}, o2 = {}, o3 = {};
#pragma unroll
  for (int r = 0; r < 8; ++r) { mrow[r] = -__builtin_inff(); lrow[r] = 0.f; }

  const float scale = 0.125f;  // 1/sqrt(64)
  const int keyEnd = qBlock + 64;

  for (int keyBase = 0; keyBase < keyEnd; keyBase += 32) {
    __syncthreads();  // previous iteration's LDS reads complete
    {
      const int key = tid >> 2;
      const int ds  = (tid & 3) * 16;
      // K block: bf16 in memory, no transform needed -> async copy to LDS.
      const unsigned short* kg =
          qkv + (rowBase + keyBase + key) * LDQ + E + h * D + ds;
      const unsigned kdst = lds_off(&ldsK[key * KP + ds]);
      async_load_b128(kdst, kg);
      async_load_b128(kdst + 16, kg + 8);
      // V block: staged transposed [d][key] through VGPRs.
      const unsigned short* vg =
          qkv + (rowBase + keyBase + key) * LDQ + 2 * E + h * D + ds;
      FragU t;
      t.q[0] = *(const U4*)vg;
      t.q[1] = *(const U4*)(vg + 8);
#pragma unroll
      for (int i = 0; i < 16; ++i)
        ldsVt[(ds + i) * VP + key] = t.us[i];
    }
    wait_asynccnt0();  // our async LDS writes landed before we signal
    __syncthreads();

    if (keyBase <= qWaveMax) {  // wave-uniform: EXEC all-ones inside
      // ---- scores: S = Q K^T, 16q x 32keys, f32 accum ----
      v8f s0 = {}, s1 = {};
      const int kst = half * 16;
#pragma unroll
      for (int dc = 0; dc < 2; ++dc) {
        const unsigned short* kr0 = &ldsK[l16 * KP + dc * 32 + kst];
        const unsigned short* kr1 = &ldsK[(l16 + 16) * KP + dc * 32 + kst];
        const v16bf kb0 = load_frag16(kr0, kr0 + 8);
        const v16bf kb1 = load_frag16(kr1, kr1 + 8);
        const v16bf qa  = dc ? qa1 : qa0;
        s0 = __builtin_amdgcn_wmma_f32_16x16x32_bf16(false, qa, false, kb0,
                                                     (short)0, s0, false, false);
        s1 = __builtin_amdgcn_wmma_f32_16x16x32_bf16(false, qa, false, kb1,
                                                     (short)0, s1, false, false);
      }

      // ---- online softmax (DPP16 row reductions, no LDS traffic) ----
      unsigned short* myP = &ldsP[wave][0];
      const int col0 = keyBase + l16;
      const int col1 = col0 + 16;
#pragma unroll
      for (int r = 0; r < 8; ++r) {
        const int qrow = qWaveBase + r + half * 8;
        float x0 = s0[r] * scale;
        float x1 = s1[r] * scale;
        if (col0 > qrow) x0 = -__builtin_inff();
        if (col1 > qrow) x1 = -__builtin_inff();
        const float rm   = reduce_max16(fmaxf(x0, x1));
        const float mnew  = fmaxf(mrow[r], rm);
        const float alpha = __expf(mrow[r] - mnew);
        const float p0 = __expf(x0 - mnew);
        const float p1 = __expf(x1 - mnew);
        const float rs = reduce_sum16(p0 + p1);
        lrow[r] = lrow[r] * alpha + rs;
        mrow[r] = mnew;
        o0[r] *= alpha; o1[r] *= alpha; o2[r] *= alpha; o3[r] *= alpha;
        const int prow = r + half * 8;
        myP[prow * VP + l16]      = f32_to_bf16(p0);
        myP[prow * VP + 16 + l16] = f32_to_bf16(p1);
      }

      // P as A fragment (wave-private LDS; LDS ops are in-order per wave)
      const v16bf pa = load_frag16(&myP[l16 * VP + aoff],
                                   &myP[l16 * VP + 8 + aoff]);

      // ---- O += P * V : 4 WMMAs over D ----
#pragma unroll
      for (int j = 0; j < 4; ++j) {
        const unsigned short* vr = &ldsVt[(j * 16 + l16) * VP + kst];
        const v16bf vb = load_frag16(vr, vr + 8);
        v8f* oacc = (j == 0) ? &o0 : (j == 1) ? &o1 : (j == 2) ? &o2 : &o3;
        *oacc = __builtin_amdgcn_wmma_f32_16x16x32_bf16(false, pa, false, vb,
                                                        (short)0, *oacc,
                                                        false, false);
      }
    }
  }

  // ---- normalize and store bf16 (row = b*S+s, col = h*64+d) ----
#pragma unroll
  for (int r = 0; r < 8; ++r) {
    const int qs = qWaveBase + r + half * 8;
    const float inv = 1.0f / lrow[r];
    unsigned short* orow = attnOut + (rowBase + qs) * E + h * D;
    orow[0 * 16 + l16] = f32_to_bf16(o0[r] * inv);
    orow[1 * 16 + l16] = f32_to_bf16(o1[r] * inv);
    orow[2 * 16 + l16] = f32_to_bf16(o2[r] * inv);
    orow[3 * 16 + l16] = f32_to_bf16(o3[r] * inv);
  }
}

// ---------------------------------------------------------------------------
extern "C" void kernel_launch(void* const* d_in, const int* in_sizes, int n_in,
                              void* d_out, int out_size, void* d_ws,
                              size_t ws_size, hipStream_t stream) {
  (void)in_sizes; (void)n_in; (void)out_size; (void)ws_size;
  const float* x     = (const float*)d_in[0];
  const float* Wqkv  = (const float*)d_in[1];
  const float* bqkv  = (const float*)d_in[2];
  const float* Wproj = (const float*)d_in[3];
  const float* bproj = (const float*)d_in[4];
  float* out = (float*)d_out;

  constexpr int B = 4, S = 1024, E = 1024;
  constexpr int M = B * S;  // 4096

  // workspace: qkv bf16 (M x 3E = 24MB) | attn bf16 (M x E = 8MB)
  unsigned short* qkv_bf  = (unsigned short*)d_ws;
  unsigned short* attn_bf = qkv_bf + (size_t)M * 3 * E;

  wmma_gemm_bias<true, true>
      <<<dim3(3 * E / 64, M / 64), dim3(256), 0, stream>>>(
          x, Wqkv, bqkv, qkv_bf, M, 3 * E, E, E, 3 * E, 3 * E);

  flash_attn_kernel<<<dim3(S / 64, 16, B), dim3(128), 0, stream>>>(qkv_bf,
                                                                   attn_bf);

  wmma_gemm_bias<false, false>
      <<<dim3(E / 64, M / 64), dim3(256), 0, stream>>>(
          attn_bf, Wproj, bproj, out, M, E, E, E, E, E);
}
